// CausalAttention_89893665506083
// MI455X (gfx1250) — matrix operand order repaired and checked by reading, not compiled, not run on hardware
//
#include <hip/hip_runtime.h>
#include <hip/hip_bf16.h>

typedef __attribute__((ext_vector_type(16))) _Float16 v16h;
typedef __attribute__((ext_vector_type(8)))  _Float16 v8h;
typedef __attribute__((ext_vector_type(4)))  _Float16 v4h;
typedef __attribute__((ext_vector_type(8)))  float    v8f;

#define NB    4
#define NS    2048
#define ND    1024
#define NBS   (NB * NS)                  // 8192 rows
#define BSD   ((size_t)NB * NS * ND)
#define SCALE 0.03125f                   // 1/sqrt(1024)

__device__ __forceinline__ v16h cat16(v8h lo, v8h hi) {
  return __builtin_shufflevector(lo, hi, 0,1,2,3,4,5,6,7,8,9,10,11,12,13,14,15);
}

// A fragment (16x32 f16): lane row = lane&15, g = lane>>4
// half h -> K = h + 8g (h<8) ; K = h + 8 + 8g (h>=8) => two contiguous 8-half runs
__device__ __forceinline__ v16h load_a32(const _Float16* rowp, int g) {
  v8h lo = *(const v8h*)(rowp + 8 * g);
  v8h hi = *(const v8h*)(rowp + 16 + 8 * g);
  return cat16(lo, hi);
}

// B fragment (32x16 f16): lane column = lane&15, g = lane>>4, half h -> K = h + 8g (h<8) ; K = h + 8 + 8g (h>=8)
// rowp points at the 32 contiguous K values belonging to this lane's column.
__device__ __forceinline__ v16h load_b32(const _Float16* rowp, int g) {
  v8h lo = *(const v8h*)(rowp + 8 * g);
  v8h hi = *(const v8h*)(rowp + 16 + 8 * g);
  return cat16(lo, hi);
}

__device__ __forceinline__ v8f wmma_f16(v16h a, v16h b, v8f c) {
  return __builtin_amdgcn_wmma_f32_16x16x32_f16(false, a, false, b, (short)0, c,
                                                false, false);
}

// ---------------------------------------------------------------------------
// Kernel 1: y = (x @ W^T + b) [* scale for Q], stored f16 into workspace.
//   z=0 -> Q row-major [row][d]    (pre-scaled by 1/sqrt(D))
//   z=1 -> K row-major [row][d]
//   z=2 -> V TRANSPOSED [d][row]   (so P@V B-fragments are contiguous loads)
// grid = (N/128, M/128, 3)  block = 256 threads (8 waves, 2x4 wave tiling)
// ---------------------------------------------------------------------------
__global__ __launch_bounds__(256)
void qkv_gemm_kernel(const float* __restrict__ x,
                     const float* __restrict__ Wq, const float* __restrict__ bq,
                     const float* __restrict__ Wk, const float* __restrict__ bk,
                     const float* __restrict__ Wv, const float* __restrict__ bv,
                     _Float16* __restrict__ qkv) {
  constexpr int LDT = 40;                       // padded row stride (halves)
  __shared__ _Float16 As[128 * LDT];
  __shared__ _Float16 Bs[128 * LDT];

  const int z = blockIdx.z;
  const float* W    = (z == 0) ? Wq : ((z == 1) ? Wk : Wv);
  const float* bias = (z == 0) ? bq : ((z == 1) ? bk : bv);
  const float scale = (z == 0) ? SCALE : 1.0f;

  const int tid  = threadIdx.x;
  const int wave = tid >> 5, lane = tid & 31;
  const int g = lane >> 4, ln = lane & 15;
  const int wm = wave & 1, wn = wave >> 1;      // 2 x 4 waves -> 64M x 32N each
  const int m0 = blockIdx.y * 128, n0 = blockIdx.x * 128;

  const int lr = tid >> 3;                      // 0..31 (row within 32-row slab)
  const int lc = (tid & 7) * 4;                 // 0,4,...,28 (K column)

  v8f acc[4][2] = {};

  for (int kt = 0; kt < ND; kt += 32) {
    #pragma unroll
    for (int i = 0; i < 4; ++i) {
      const int row = lr + 32 * i;
      const float4 fa = *(const float4*)(x + (size_t)(m0 + row) * ND + kt + lc);
      const float4 fb = *(const float4*)(W + (size_t)(n0 + row) * ND + kt + lc);
      v4h ha = {(_Float16)fa.x, (_Float16)fa.y, (_Float16)fa.z, (_Float16)fa.w};
      v4h hb = {(_Float16)fb.x, (_Float16)fb.y, (_Float16)fb.z, (_Float16)fb.w};
      *(v4h*)(As + row * LDT + lc) = ha;
      *(v4h*)(Bs + row * LDT + lc) = hb;
    }
    __syncthreads();

    v16h af[4], bf[2];
    #pragma unroll
    for (int mt = 0; mt < 4; ++mt)
      af[mt] = load_a32(As + (wm * 64 + mt * 16 + ln) * LDT, g);
    #pragma unroll
    for (int nt = 0; nt < 2; ++nt)
      bf[nt] = load_b32(Bs + (wn * 32 + nt * 16 + ln) * LDT, g);

    #pragma unroll
    for (int mt = 0; mt < 4; ++mt)
      #pragma unroll
      for (int nt = 0; nt < 2; ++nt)
        acc[mt][nt] = wmma_f16(af[mt], bf[nt], acc[mt][nt]);
    __syncthreads();
  }

  if (z < 2) {
    _Float16* y = qkv + (size_t)z * BSD;        // row-major Q or K
    #pragma unroll
    for (int mt = 0; mt < 4; ++mt)
      #pragma unroll
      for (int nt = 0; nt < 2; ++nt) {
        const int col = n0 + wn * 32 + nt * 16 + ln;
        const float bcol = bias[col];
        #pragma unroll
        for (int r = 0; r < 8; ++r) {
          const int row = m0 + wm * 64 + mt * 16 + r + 8 * g;
          y[(size_t)row * ND + col] = (_Float16)((acc[mt][nt][r] + bcol) * scale);
        }
      }
  } else {
    // V transposed: Vt[d][row]. A lane's 8 accumulator rows are consecutive
    // (base + 8g + 0..7) at a fixed column d -> single contiguous v8h store.
    _Float16* vt = qkv + 2 * BSD;
    #pragma unroll
    for (int mt = 0; mt < 4; ++mt)
      #pragma unroll
      for (int nt = 0; nt < 2; ++nt) {
        const int col = n0 + wn * 32 + nt * 16 + ln;      // d index
        const float bcol = bias[col];
        const int rowbase = m0 + wm * 64 + mt * 16 + 8 * g;
        v8h pack;
        #pragma unroll
        for (int r = 0; r < 8; ++r)
          pack[r] = (_Float16)(acc[mt][nt][r] + bcol);
        *(v8h*)(vt + (size_t)col * NBS + rowbase) = pack;
      }
  }
}

// ---------------------------------------------------------------------------
// Kernel 2: flash attention. grid = B * S/32 blocks, 256 threads.
// Wave w owns d-slice [w*128, w*128+128); all 32 query rows per wave.
// ---------------------------------------------------------------------------
__global__ __launch_bounds__(256)
void flash_attn_kernel(const _Float16* __restrict__ qkv,
                       float* __restrict__ out) {
  __shared__ float    s_part[8 * 1024];       // per-wave partial 32x32 scores
  __shared__ float    s_full[1024];           // reduced 32x32 scores
  __shared__ _Float16 pstage[8 * 32 * 40];    // per-wave P staging (padded)

  const int tid = threadIdx.x;
  const int w = tid >> 5, lane = tid & 31, g = lane >> 4, n = lane & 15;
  const int b  = blockIdx.x >> 6;             // 64 q-blocks per batch
  const int q0 = (blockIdx.x & 63) * 32;
  const _Float16* qp  = qkv;
  const _Float16* kp  = qkv + BSD;
  const _Float16* vtp = qkv + 2 * BSD;        // Vt[d][row]
  const int dsb = w * 128;
  const size_t rowbase = (size_t)b * NS;

  // Resident Q fragments: 2 row-tiles x 4 K-chunks (d) per wave.
  v16h aq[2][4];
  #pragma unroll
  for (int rt = 0; rt < 2; ++rt)
    #pragma unroll
    for (int dc = 0; dc < 4; ++dc)
      aq[rt][dc] =
          load_a32(qp + (rowbase + q0 + rt * 16 + n) * ND + dsb + dc * 32, g);

  v8f o[2][8] = {};
  float mrow[2][8], lrow[2][8];
  #pragma unroll
  for (int rt = 0; rt < 2; ++rt)
    #pragma unroll
    for (int r = 0; r < 8; ++r) { mrow[rt][r] = -1e30f; lrow[rt][r] = 0.0f; }

  const int ktmax = q0 >> 5;
  for (int kt = 0; kt <= ktmax; ++kt) {
    const int kb = kt * 32;

    // ---- partial scores over this wave's 128-wide d-slice ----
    v8f s[2][2];
    #pragma unroll
    for (int rt = 0; rt < 2; ++rt)
      #pragma unroll
      for (int ct = 0; ct < 2; ++ct) {
        v8f a = {};
        #pragma unroll
        for (int dc = 0; dc < 4; ++dc) {
          v16h bk = load_b32(
              kp + (rowbase + kb + ct * 16 + n) * ND + dsb + dc * 32, g);
          a = wmma_f16(aq[rt][dc], bk, a);
        }
        s[rt][ct] = a;
      }

    // ---- cross-wave reduction of scores through LDS ----
    float* myp = s_part + w * 1024;
    #pragma unroll
    for (int rt = 0; rt < 2; ++rt)
      #pragma unroll
      for (int ct = 0; ct < 2; ++ct)
        #pragma unroll
        for (int r = 0; r < 8; ++r)
          myp[((rt * 2 + ct) * 8 + r) * 32 + lane] = s[rt][ct][r];
    __syncthreads();
    if (w < 4) {                                // wave w reduces tile w
      #pragma unroll
      for (int r = 0; r < 8; ++r) {
        float t = 0.0f;
        #pragma unroll
        for (int ow = 0; ow < 8; ++ow)
          t += s_part[ow * 1024 + (w * 8 + r) * 32 + lane];
        s_full[(w * 8 + r) * 32 + lane] = t;
      }
    }
    __syncthreads();
    #pragma unroll
    for (int rt = 0; rt < 2; ++rt)
      #pragma unroll
      for (int ct = 0; ct < 2; ++ct)
        #pragma unroll
        for (int r = 0; r < 8; ++r)
          s[rt][ct][r] = s_full[((rt * 2 + ct) * 8 + r) * 32 + lane];

    // ---- online softmax (replicated across waves) ----
    const bool diag = (kt == ktmax);
    #pragma unroll
    for (int rt = 0; rt < 2; ++rt) {
      #pragma unroll
      for (int r = 0; r < 8; ++r) {
        const int qrow = q0 + rt * 16 + r + 8 * g;
        float s0 = s[rt][0][r], s1 = s[rt][1][r];
        if (diag) {
          if (kb + n > qrow)      s0 = -1e30f;
          if (kb + 16 + n > qrow) s1 = -1e30f;
        }
        float mx = fmaxf(s0, s1);
        #pragma unroll
        for (int off = 1; off < 16; off <<= 1)
          mx = fmaxf(mx, __shfl_xor(mx, off, 32));
        const float mnew  = fmaxf(mrow[rt][r], mx);
        const float alpha = __expf(mrow[rt][r] - mnew);
        const float p0 = __expf(s0 - mnew);
        const float p1 = __expf(s1 - mnew);
        float rs = p0 + p1;
        #pragma unroll
        for (int off = 1; off < 16; off <<= 1)
          rs += __shfl_xor(rs, off, 32);
        lrow[rt][r] = lrow[rt][r] * alpha + rs;
        mrow[rt][r] = mnew;
        #pragma unroll
        for (int dt = 0; dt < 8; ++dt) o[rt][dt][r] *= alpha;
        _Float16* pr = pstage + w * 1280 + (rt * 16 + r + 8 * g) * 40;
        pr[n]      = (_Float16)p0;
        pr[16 + n] = (_Float16)p1;
      }
    }
    __syncthreads();   // also fences P staging (wave-private region)

    // ---- O += P @ V (this wave's d-slice), V fragments from Vt contiguous ----
    v16h ap[2];
    ap[0] = load_a32(pstage + w * 1280 + n * 40, g);
    ap[1] = load_a32(pstage + w * 1280 + (16 + n) * 40, g);
    #pragma unroll
    for (int dt = 0; dt < 8; ++dt) {
      const int d = dsb + dt * 16 + n;
      v16h bfv = load_b32(vtp + (size_t)d * NBS + rowbase + kb, g);
      o[0][dt] = wmma_f16(ap[0], bfv, o[0][dt]);
      o[1][dt] = wmma_f16(ap[1], bfv, o[1][dt]);
    }
    // prefetch next K tile rows for this wave's slice
    if (kt < ktmax)
      __builtin_prefetch(kp + (rowbase + kb + 32 + n) * ND + dsb, 0, 1);
  }

  // ---- epilogue: O / l -> fp32 output ----
  #pragma unroll
  for (int rt = 0; rt < 2; ++rt)
    #pragma unroll
    for (int r = 0; r < 8; ++r) {
      const float inv = 1.0f / lrow[rt][r];
      const size_t row = rowbase + q0 + rt * 16 + r + 8 * g;
      #pragma unroll
      for (int dt = 0; dt < 8; ++dt)
        out[row * ND + dsb + dt * 16 + n] = o[rt][dt][r] * inv;
    }
}

extern "C" void kernel_launch(void* const* d_in, const int* in_sizes, int n_in,
                              void* d_out, int out_size, void* d_ws,
                              size_t ws_size, hipStream_t stream) {
  const float* x  = (const float*)d_in[0];
  const float* Wq = (const float*)d_in[1];
  const float* bq = (const float*)d_in[2];
  const float* Wk = (const float*)d_in[3];
  const float* bk = (const float*)d_in[4];
  const float* Wv = (const float*)d_in[5];
  const float* bv = (const float*)d_in[6];
  _Float16* qkv = (_Float16*)d_ws;            // Q, K row-major + Vt : 48 MB f16
  float* out = (float*)d_out;

  dim3 g1(ND / 128, NBS / 128, 3);            // (8, 64, 3)
  qkv_gemm_kernel<<<g1, 256, 0, stream>>>(x, Wq, bq, Wk, bk, Wv, bv, qkv);

  dim3 g2(NB * (NS / 32));                    // 256 blocks
  flash_attn_kernel<<<g2, 256, 0, stream>>>(qkv, out);
}
